// LGCN_mlp_AGF_9706626089565
// MI455X (gfx1250) — compile-verified, matrix operand order checked by reading, not run on hardware
//
#include <hip/hip_runtime.h>
#include <hip/hip_bf16.h>
#include <stdint.h>

#define N_NODES 100000
#define D_IN    128
#define K_HOPS  8
#define FAN1    (D_IN * (K_HOPS + 1))   // 1152
#define HID     512
#define OUT_DIM 32
#define E_EDGES 1600000
#define BN_EPS  1e-5f
#define DEG_EPS 1e-8f
#define ROW_TILES (N_NODES / 16)        // 6250

typedef __bf16        v16bf  __attribute__((ext_vector_type(16)));
typedef float         v8f    __attribute__((ext_vector_type(8)));
typedef unsigned int  u32x4  __attribute__((ext_vector_type(4)));
typedef float         f32x4  __attribute__((ext_vector_type(4)));

union FragAB { v16bf v; u32x4 q[2]; };

__device__ __forceinline__ unsigned short f2bf(float f) {
  unsigned int u = __float_as_uint(f);
  u += 0x7fffu + ((u >> 16) & 1u);      // round-to-nearest-even
  return (unsigned short)(u >> 16);
}

// ---------------------------------------------------------------- utility
__global__ void k_zero(float* __restrict__ p, long n) {
  long i = (long)blockIdx.x * blockDim.x + threadIdx.x;
  long stride = (long)gridDim.x * blockDim.x;
  for (; i < n; i += stride) p[i] = 0.0f;
}

__global__ void k_cast_f32_bf16(const float* __restrict__ s,
                                unsigned short* __restrict__ d, long n) {
  long i = (long)blockIdx.x * blockDim.x + threadIdx.x;
  if (i < n) d[i] = f2bf(s[i]);
}

// scatter hop-k result (f32 [N,128]) into column block k of h_bf16 [N,1152]
__global__ void k_cast_block(const float* __restrict__ s,
                             unsigned short* __restrict__ h, int kblk) {
  long i = (long)blockIdx.x * blockDim.x + threadIdx.x;
  if (i < (long)N_NODES * D_IN) {
    long n = i >> 7;          // / 128
    int  d = (int)(i & 127);
    h[n * FAN1 + (long)kblk * D_IN + d] = f2bf(s[i]);
  }
}

// ---------------------------------------------------------------- graph part
__global__ void k_edge(const float* __restrict__ coor,
                       const int* __restrict__ row, const int* __restrict__ col,
                       const float* __restrict__ theta_p, const int* __restrict__ m_p,
                       float* __restrict__ w, float* __restrict__ deg) {
  int e = blockIdx.x * blockDim.x + threadIdx.x;
  if (e >= E_EDGES) return;
  int r = row[e], c = col[e];
  float dx = coor[2 * r]     - coor[2 * c];
  float dy = coor[2 * r + 1] - coor[2 * c + 1];
  float t  = theta_p[0] / (float)m_p[0];
  float ww = __expf(-(dx * dx + dy * dy) * t);
  w[e] = ww;
  atomicAdd(deg + r, ww);
}

__global__ void k_norm(float* __restrict__ w, const float* __restrict__ deg,
                       const int* __restrict__ row) {
  int e = blockIdx.x * blockDim.x + threadIdx.x;
  if (e >= E_EDGES) return;
  w[e] = w[e] / (deg[row[e]] + DEG_EPS);
}

// one wave (32 lanes x float4) per edge: gather src row, scale, scatter-add
__global__ __launch_bounds__(256) void k_scatter(const float* __restrict__ src,
                                                 float* __restrict__ dst,
                                                 const int* __restrict__ row,
                                                 const int* __restrict__ col,
                                                 const float* __restrict__ norm) {
  int e = blockIdx.x * 8 + (threadIdx.x >> 5);
  if (e >= E_EDGES) return;
  int lane = threadIdx.x & 31;
  int r = row[e], c = col[e];
  float w = norm[e];
  f32x4 v = *(const f32x4*)(src + (long)c * D_IN + lane * 4);
  float* d = dst + (long)r * D_IN + lane * 4;
  atomicAdd(d + 0, v.x * w);
  atomicAdd(d + 1, v.y * w);
  atomicAdd(d + 2, v.z * w);
  atomicAdd(d + 3, v.w * w);
}

// ---------------------------------------------------------------- fc1 (WMMA)
// block = 8 waves; block -> one 16-row tile; wave w -> cols [w*64, w*64+64)
__global__ __launch_bounds__(256) void k_gemm1(
    const unsigned short* __restrict__ A,    // h   bf16 [N,1152]
    const unsigned short* __restrict__ B,    // W1  bf16 [512,1152]
    const float* __restrict__ b1,
    unsigned short* __restrict__ act,        // out bf16 [N,512]
    float* __restrict__ colsum, float* __restrict__ colsumsq) {
  const int tile = blockIdx.x;
  const int wave = threadIdx.x >> 5;
  const int lane = threadIdx.x & 31;
  const int m  = lane & 15;
  const int hh = lane >> 4;
  const int colbase = wave * 64;

  const unsigned short* arow = A + (size_t)(tile * 16 + m) * FAN1;
  v8f zero = {0.f, 0.f, 0.f, 0.f, 0.f, 0.f, 0.f, 0.f};
  v8f acc[4] = {zero, zero, zero, zero};

  for (int k = 0; k < FAN1; k += 32) {
    FragAB a;
    a.q[0] = *(const u32x4*)(arow + k + 8 * hh);
    a.q[1] = *(const u32x4*)(arow + k + 16 + 8 * hh);
#pragma unroll
    for (int t = 0; t < 4; ++t) {
      const unsigned short* brow = B + (size_t)(colbase + t * 16 + m) * FAN1;
      FragAB b;
      b.q[0] = *(const u32x4*)(brow + k + 8 * hh);
      b.q[1] = *(const u32x4*)(brow + k + 16 + 8 * hh);
      acc[t] = __builtin_amdgcn_wmma_f32_16x16x32_bf16(
          false, a.v, false, b.v, (short)0, acc[t], false, false);
    }
  }

#pragma unroll
  for (int t = 0; t < 4; ++t) {
    int col = colbase + t * 16 + m;
    float bias = b1[col];
    float s1 = 0.f, s2 = 0.f;
#pragma unroll
    for (int r = 0; r < 8; ++r) {
      float x = acc[t][r] + bias;
      x = (x > 0.f) ? x : 0.2f * x;            // leaky_relu(0.2)
      int orow = tile * 16 + r + 8 * hh;
      act[(size_t)orow * HID + col] = f2bf(x);
      s1 += x;
      s2 += x * x;
    }
    atomicAdd(colsum + col, s1);
    atomicAdd(colsumsq + col, s2);
  }
}

// ---------------------------------------------------------------- BN fold
// 32 blocks (one per output o), 512 threads (one per channel c)
__global__ __launch_bounds__(512) void k_bnfold(
    const float* __restrict__ colsum, const float* __restrict__ colsumsq,
    const float* __restrict__ gamma, const float* __restrict__ beta,
    const float* __restrict__ W2, const float* __restrict__ b2,
    unsigned short* __restrict__ W2f, float* __restrict__ b2f) {
  __shared__ float red[512];
  int o = blockIdx.x;
  int c = threadIdx.x;
  const float inv = 1.0f / (float)N_NODES;
  float mu  = colsum[c] * inv;
  float var = colsumsq[c] * inv - mu * mu;
  float s = gamma[c] * rsqrtf(var + BN_EPS);
  float t = beta[c] - mu * s;
  float w = W2[o * HID + c];
  W2f[o * HID + c] = f2bf(w * s);
  red[c] = w * t;
  __syncthreads();
  for (int st = 256; st > 0; st >>= 1) {
    if (c < st) red[c] += red[c + st];
    __syncthreads();
  }
  if (c == 0) b2f[o] = b2[o] + red[0];
}

// ---------------------------------------------------------------- fc2 (WMMA)
// one wave per 16-row tile, 2 col tiles (OUT=32); 8 tiles per block
__global__ __launch_bounds__(256) void k_gemm2(
    const unsigned short* __restrict__ A,    // act bf16 [N,512]
    const unsigned short* __restrict__ B,    // W2f bf16 [32,512]
    const float* __restrict__ b2f,
    float* __restrict__ out) {               // [N,32] f32
  int tile = blockIdx.x * 8 + (threadIdx.x >> 5);
  if (tile >= ROW_TILES) return;
  int lane = threadIdx.x & 31;
  int m  = lane & 15;
  int hh = lane >> 4;
  const unsigned short* arow = A + (size_t)(tile * 16 + m) * HID;
  v8f zero = {0.f, 0.f, 0.f, 0.f, 0.f, 0.f, 0.f, 0.f};
  v8f acc[2] = {zero, zero};

  for (int k = 0; k < HID; k += 32) {
    FragAB a;
    a.q[0] = *(const u32x4*)(arow + k + 8 * hh);
    a.q[1] = *(const u32x4*)(arow + k + 16 + 8 * hh);
#pragma unroll
    for (int t = 0; t < 2; ++t) {
      const unsigned short* brow = B + (size_t)(t * 16 + m) * HID;
      FragAB b;
      b.q[0] = *(const u32x4*)(brow + k + 8 * hh);
      b.q[1] = *(const u32x4*)(brow + k + 16 + 8 * hh);
      acc[t] = __builtin_amdgcn_wmma_f32_16x16x32_bf16(
          false, a.v, false, b.v, (short)0, acc[t], false, false);
    }
  }

#pragma unroll
  for (int t = 0; t < 2; ++t) {
    int col = t * 16 + m;
    float bias = b2f[col];
#pragma unroll
    for (int r = 0; r < 8; ++r)
      out[(size_t)(tile * 16 + r + 8 * hh) * OUT_DIM + col] = acc[t][r] + bias;
  }
}

// ---------------------------------------------------------------- launch
extern "C" void kernel_launch(void* const* d_in, const int* in_sizes, int n_in,
                              void* d_out, int out_size, void* d_ws, size_t ws_size,
                              hipStream_t stream) {
  (void)in_sizes; (void)n_in; (void)out_size; (void)ws_size;
  const float* feature = (const float*)d_in[0];
  const float* coor    = (const float*)d_in[1];
  const float* theta   = (const float*)d_in[2];
  const float* W1      = (const float*)d_in[3];
  const float* b1      = (const float*)d_in[4];
  const float* gamma   = (const float*)d_in[5];
  const float* beta    = (const float*)d_in[6];
  const float* W2      = (const float*)d_in[7];
  const float* b2      = (const float*)d_in[8];
  const int*   eidx    = (const int*)d_in[9];
  const int*   m_p     = (const int*)d_in[10];
  const int* row = eidx;             // dst
  const int* col = eidx + E_EDGES;   // src
  float* out = (float*)d_out;

  // ---- workspace layout (all 256B-aligned) ----
  char* ws = (char*)d_ws;
  constexpr size_t SZ_X    = (size_t)N_NODES * D_IN * sizeof(float);           // 51,200,000
  constexpr size_t SZ_H    = (size_t)N_NODES * FAN1 * sizeof(unsigned short);  // 230,400,000
  constexpr size_t SZ_ACT  = (size_t)N_NODES * HID * sizeof(unsigned short);   // 102,400,000
  constexpr size_t SZ_W1B  = (size_t)HID * FAN1 * sizeof(unsigned short);      // 1,179,648
  constexpr size_t SZ_W2F  = (size_t)OUT_DIM * HID * sizeof(unsigned short);   // 32,768
  constexpr size_t SZ_DEGA = (((size_t)N_NODES * sizeof(float)) + 255) & ~(size_t)255;

  constexpr size_t OFF_XA  = 0;
  constexpr size_t OFF_XB  = OFF_XA + SZ_X;
  constexpr size_t OFF_H   = OFF_XB + SZ_X;
  constexpr size_t OFF_ACT = OFF_H + SZ_H;
  constexpr size_t OFF_W1B = OFF_ACT + SZ_ACT;
  constexpr size_t OFF_W2F = OFF_W1B + SZ_W1B;
  constexpr size_t OFF_B2F = OFF_W2F + SZ_W2F;
  constexpr size_t OFF_DEG = OFF_B2F + 256;
  constexpr size_t OFF_CS  = OFF_DEG + SZ_DEGA;
  constexpr size_t OFF_CSQ = OFF_CS + HID * sizeof(float);
  constexpr size_t OFF_WN  = OFF_CSQ + HID * sizeof(float);

  float*          xa    = (float*)(ws + OFF_XA);
  float*          xb    = (float*)(ws + OFF_XB);
  unsigned short* hbf   = (unsigned short*)(ws + OFF_H);
  unsigned short* act   = (unsigned short*)(ws + OFF_ACT);
  unsigned short* W1b   = (unsigned short*)(ws + OFF_W1B);
  unsigned short* W2f   = (unsigned short*)(ws + OFF_W2F);
  float*          b2f   = (float*)(ws + OFF_B2F);
  float*          deg   = (float*)(ws + OFF_DEG);
  float*          csum  = (float*)(ws + OFF_CS);
  float*          csumq = (float*)(ws + OFF_CSQ);
  float*          wnorm = (float*)(ws + OFF_WN);

  // zero deg + colsum + colsumsq (contiguous region)
  long zn = (long)((SZ_DEGA + 2 * HID * sizeof(float)) / sizeof(float));
  k_zero<<<256, 256, 0, stream>>>(deg, zn);

  // W1 -> bf16
  k_cast_f32_bf16<<<(HID * FAN1 + 255) / 256, 256, 0, stream>>>(W1, W1b, (long)HID * FAN1);

  // edge weights + degree, then normalize
  k_edge<<<(E_EDGES + 255) / 256, 256, 0, stream>>>(coor, row, col, theta, m_p, wnorm, deg);
  k_norm<<<(E_EDGES + 255) / 256, 256, 0, stream>>>(wnorm, deg, row);

  // hop 0 = input feature into h block 0
  k_cast_block<<<(N_NODES * D_IN + 255) / 256, 256, 0, stream>>>(feature, hbf, 0);

  // K sequential propagations (ping-pong xa/xb)
  const float* src = feature;
  for (int k = 1; k <= K_HOPS; ++k) {
    float* dst = (k & 1) ? xa : xb;
    k_zero<<<4096, 256, 0, stream>>>(dst, (long)N_NODES * D_IN);
    k_scatter<<<E_EDGES / 8, 256, 0, stream>>>(src, dst, row, col, wnorm);
    k_cast_block<<<(N_NODES * D_IN + 255) / 256, 256, 0, stream>>>(dst, hbf, k);
    src = dst;
  }

  // fc1 + leaky_relu + BN-stat accumulation (WMMA bf16)
  k_gemm1<<<ROW_TILES, 256, 0, stream>>>(hbf, W1b, b1, act, csum, csumq);

  // fold batch-norm into fc2 weights/bias
  k_bnfold<<<OUT_DIM, 512, 0, stream>>>(csum, csumq, gamma, beta, W2, b2, W2f, b2f);

  // fc2 (WMMA bf16) -> f32 output
  k_gemm2<<<(ROW_TILES + 7) / 8, 256, 0, stream>>>(act, W2f, b2f, out);
}
